// TemporalPatternAttentionMechanism_9259949490413
// MI455X (gfx1250) — matrix operand order repaired and checked by reading, not compiled, not run on hardware
//
#include <hip/hip_runtime.h>

typedef __attribute__((ext_vector_type(16))) _Float16 v16h;
typedef __attribute__((ext_vector_type(8)))  _Float16 v8h;
typedef __attribute__((ext_vector_type(8)))  float    v8f;

#define BATCH 2048
#define SEQ   512
#define ISZ   256
#define FSZ   256

// ---------------------------------------------------------------------------
// Workspace layout (bytes):
//   [0, 262144)                : Wc converted to f16, [F][S] row-major
//   [262144, 262144+2097152)   : w = hidden @ W1^T, [B][F] f32
//   [2359296, 2359296+2052)    : u[0..511] = Wfc@W2 (f32), u[512] = Wfc.b2+bfc
// ---------------------------------------------------------------------------

__global__ void k_convert_wc(const float* __restrict__ wc, _Float16* __restrict__ wch) {
    int i = blockIdx.x * blockDim.x + threadIdx.x;   // FSZ*SEQ threads total
    wch[i] = (_Float16)wc[i];                        // v_cvt_f16_f32 (RNE)
}

__global__ void k_compute_w(const float* __restrict__ hidden, const float* __restrict__ W1,
                            float* __restrict__ wout) {
    __shared__ float h[ISZ];
    const int b = blockIdx.x, t = threadIdx.x;
    h[t] = hidden[(size_t)b * ISZ + t];
    __syncthreads();
    const float4* w1r = (const float4*)(W1 + (size_t)t * ISZ);
    const float4* hp  = (const float4*)h;
    float acc = 0.f;
#pragma unroll 8
    for (int k = 0; k < ISZ / 4; ++k) {
        float4 a = w1r[k], v = hp[k];
        acc += a.x * v.x + a.y * v.y + a.z * v.z + a.w * v.w;
    }
    wout[(size_t)b * FSZ + t] = acc;
}

__global__ void k_compute_u(const float* __restrict__ Wfc, const float* __restrict__ W2,
                            const float* __restrict__ b2, const float* __restrict__ bfc,
                            float* __restrict__ u) {
    const int k = threadIdx.x;   // 512 threads
    float acc = 0.f;
    for (int j = 0; j < ISZ; ++j) acc += Wfc[j] * W2[(size_t)j * (ISZ + FSZ) + k];
    u[k] = acc;
    if (k == 0) {
        float c = bfc[0];
        for (int j = 0; j < ISZ; ++j) c += Wfc[j] * b2[j];
        u[512] = c;
    }
}

// ---------------------------------------------------------------------------
// Main fused kernel: one 512-thread WG (16 waves) per batch element.
// Wave (r,c) owns the 64x64 block at rows [64r,64r+64), cols [64c,64c+64)
// of M_b = Wc @ inp_b, as a 4x4 grid of 16x16 WMMA tiles (v8f accumulators).
// Register double-buffering: the inp chunk for step kk+32 is loaded into
// VGPRs while the WMMAs / staging of step kk execute; f32->f16 packing uses
// v_cvt_pk_f16_f32 (one VALU op per packed pair).
// ---------------------------------------------------------------------------
__global__ __launch_bounds__(512, 1) void k_tpa_main(
    const float*    __restrict__ inp,     // [B][S][I] f32
    const float*    __restrict__ hidden,  // [B][I]
    const float*    __restrict__ bc,      // [F]
    const _Float16* __restrict__ wch,     // [F][S] f16
    const float*    __restrict__ wv,      // [B][F] = hidden @ W1^T
    const float*    __restrict__ u,       // [513]
    float*          __restrict__ out)     // [B]
{
    const int b    = blockIdx.x;
    const int t    = threadIdx.x;
    const int lane = t & 31;
    const int wave = t >> 5;
    const int r    = wave >> 2;        // row-block 0..3
    const int c    = wave & 3;         // col-block 0..3
    const int l16  = lane & 15;
    const int hsel = lane >> 4;        // 0 for lanes 0-15, 1 for lanes 16-31

    // Transposed K-chunk of inp_b: bT[n][k], n=0..255 (I), k=0..31 (K chunk),
    // row stride 40 halves (80 B): 16B-aligned b128 reads, conflict-free stores.
    __shared__ _Float16 bT[256 * 40];
    __shared__ float w_lds[FSZ];
    __shared__ float s_lds[FSZ];
    __shared__ float a_lds[FSZ];
    __shared__ float v_lds[FSZ];
    __shared__ float out_acc;

    if (t < 256) { w_lds[t] = wv[(size_t)b * FSZ + t]; s_lds[t] = 0.f; v_lds[t] = 0.f; }
    if (t == 0) out_acc = u[512];

    v8f acc[4][4];
#pragma unroll
    for (int i = 0; i < 4; ++i)
#pragma unroll
        for (int j = 0; j < 4; ++j) acc[i][j] = {};

    const float* inpb = inp + (size_t)b * (SEQ * ISZ);

    // Per-thread staging coordinates: unit = t + uu*512;
    // cg = unit>>4 (col group of 4), rp = unit&15 (row pair).
    int cg_[2], rp_[2];
#pragma unroll
    for (int uu = 0; uu < 2; ++uu) {
        int unit = t + uu * 512;
        cg_[uu] = unit >> 4;
        rp_[uu] = unit & 15;
    }

    float4 pf0[2], pf1[2];   // prefetched rows (row, row+1) x 4 cols per unit

    // prefetch chunk kk = 0
#pragma unroll
    for (int uu = 0; uu < 2; ++uu) {
        int row = rp_[uu] * 2;
        pf0[uu] = *(const float4*)(inpb + (size_t)row * ISZ + cg_[uu] * 4);
        pf1[uu] = *(const float4*)(inpb + (size_t)(row + 1) * ISZ + cg_[uu] * 4);
    }

    for (int kk = 0; kk < SEQ; kk += 32) {
        __syncthreads();
        // ---- store prefetched 32x256 chunk -> f16, transposed into LDS ------
        // bT[n][klocal] pair packed with v_cvt_pk_f16_f32 (low = row, high = row+1).
#pragma unroll
        for (int uu = 0; uu < 2; ++uu) {
            float lo[4] = {pf0[uu].x, pf0[uu].y, pf0[uu].z, pf0[uu].w};
            float hi[4] = {pf1[uu].x, pf1[uu].y, pf1[uu].z, pf1[uu].w};
            unsigned int* base = (unsigned int*)&bT[0];
            int klocal = rp_[uu] * 2;
#pragma unroll
            for (int j = 0; j < 4; ++j) {
                auto pk = __builtin_amdgcn_cvt_pkrtz(lo[j], hi[j]);  // v_cvt_pk_f16_f32
                unsigned int idx_h = (unsigned)(cg_[uu] * 4 + j) * 40 + klocal;  // halves
                base[idx_h >> 1] = __builtin_bit_cast(unsigned int, pk);
            }
        }
        __syncthreads();

        // ---- issue prefetch of the NEXT chunk (in flight under the WMMAs) ---
        if (kk + 32 < SEQ) {
#pragma unroll
            for (int uu = 0; uu < 2; ++uu) {
                int row = kk + 32 + rp_[uu] * 2;
                pf0[uu] = *(const float4*)(inpb + (size_t)row * ISZ + cg_[uu] * 4);
                pf1[uu] = *(const float4*)(inpb + (size_t)(row + 1) * ISZ + cg_[uu] * 4);
            }
        }

        // ---- B operands (32x16 f16 tiles) from LDS --------------------------
        // B k order used here (contiguous, same as A): lanes 0-15 = col l16 with K 0..15; lanes 16-31 = K 16..31.
        v16h bmat[4];
#pragma unroll
        for (int j = 0; j < 4; ++j) {
            int n = c * 64 + j * 16 + l16;
            const v8h* p = (const v8h*)&bT[n * 40 + hsel * 16];
            v8h b0 = p[0];
            v8h b1 = p[1];
            bmat[j] = __builtin_shufflevector(b0, b1, 0,1,2,3,4,5,6,7,8,9,10,11,12,13,14,15);
        }
        // ---- A operands (16x32 f16 tiles) from L2-resident Wc(f16) ----------
        // A k order used here (contiguous, same as B): lane<16 row l16 K 0..15; lane>=16 K 16..31.
        v16h amat[4];
#pragma unroll
        for (int i = 0; i < 4; ++i) {
            int m = r * 64 + i * 16 + l16;
            const _Float16* ap = wch + (size_t)m * SEQ + kk + hsel * 16;
            v8h a0 = *(const v8h*)ap;
            v8h a1 = *(const v8h*)(ap + 8);
            amat[i] = __builtin_shufflevector(a0, a1, 0,1,2,3,4,5,6,7,8,9,10,11,12,13,14,15);
        }
#pragma unroll
        for (int i = 0; i < 4; ++i) {
#pragma unroll
            for (int j = 0; j < 4; ++j) {
                acc[i][j] = __builtin_amdgcn_wmma_f32_16x16x32_f16(
                    false, amat[i], false, bmat[j], (short)0, acc[i][j], false, false);
            }
        }
    }

    // ---- epilogue: M += bc[row];  s[x] = sum_y M[x,y]*w[y] ------------------
    // C/D layout: lanes 0-15 hold cols, element v = row v (lanes 16-31: row v+8).
    float wy[4];
#pragma unroll
    for (int j = 0; j < 4; ++j) wy[j] = w_lds[c * 64 + j * 16 + l16];

#pragma unroll
    for (int i = 0; i < 4; ++i) {
        int xbase = r * 64 + i * 16 + hsel * 8;
#pragma unroll
        for (int v = 0; v < 8; ++v) {
            float bcv = bc[xbase + v];
            float p = 0.f;
#pragma unroll
            for (int j = 0; j < 4; ++j) {
                float mval = acc[i][j][v] + bcv;
                acc[i][j][v] = mval;          // keep biased M for the v-phase
                p += mval * wy[j];
            }
            // reduce across the 16 lanes that share this row
            p += __shfl_xor(p, 1, 32);
            p += __shfl_xor(p, 2, 32);
            p += __shfl_xor(p, 4, 32);
            p += __shfl_xor(p, 8, 32);
            if (l16 == 0) atomicAdd(&s_lds[xbase + v], p);
        }
    }
    __syncthreads();
    if (t < 256) a_lds[t] = 1.0f / (1.0f + __expf(-s_lds[t]));
    __syncthreads();

    // ---- v[y] = sum_x a[x] * M[x,y] ----------------------------------------
    float vp[4] = {0.f, 0.f, 0.f, 0.f};
#pragma unroll
    for (int i = 0; i < 4; ++i) {
        int xbase = r * 64 + i * 16 + hsel * 8;
#pragma unroll
        for (int v = 0; v < 8; ++v) {
            float av = a_lds[xbase + v];
#pragma unroll
            for (int j = 0; j < 4; ++j) vp[j] += av * acc[i][j][v];
        }
    }
#pragma unroll
    for (int j = 0; j < 4; ++j) {
        float pv = vp[j] + __shfl_xor(vp[j], 16, 32);   // combine row-halves
        if (hsel == 0) atomicAdd(&v_lds[c * 64 + j * 16 + l16], pv);
    }
    __syncthreads();

    // ---- out_b = u1.hidden_b + u2.v_b + c0 ----------------------------------
    if (t < 256) {
        float contrib = u[t] * hidden[(size_t)b * ISZ + t] + u[256 + t] * v_lds[t];
        atomicAdd(&out_acc, contrib);
    }
    __syncthreads();
    if (t == 0) out[b] = out_acc;
}

// ---------------------------------------------------------------------------
extern "C" void kernel_launch(void* const* d_in, const int* in_sizes, int n_in,
                              void* d_out, int out_size, void* d_ws, size_t ws_size,
                              hipStream_t stream) {
    const float* hidden = (const float*)d_in[0];
    const float* inp    = (const float*)d_in[1];
    const float* W1     = (const float*)d_in[2];
    const float* Wc     = (const float*)d_in[3];
    const float* bc     = (const float*)d_in[4];
    const float* W2     = (const float*)d_in[5];
    const float* b2     = (const float*)d_in[6];
    const float* Wfc    = (const float*)d_in[7];
    const float* bfc    = (const float*)d_in[8];
    float* out = (float*)d_out;

    _Float16* ws_wch = (_Float16*)d_ws;
    float* ws_w = (float*)((char*)d_ws + (size_t)FSZ * SEQ * 2);
    float* ws_u = (float*)((char*)d_ws + (size_t)FSZ * SEQ * 2 + (size_t)BATCH * FSZ * 4);

    k_convert_wc<<<(FSZ * SEQ) / 256, 256, 0, stream>>>(Wc, ws_wch);
    k_compute_w<<<BATCH, 256, 0, stream>>>(hidden, W1, ws_w);
    k_compute_u<<<1, 512, 0, stream>>>(Wfc, W2, b2, bfc, ws_u);
    k_tpa_main<<<BATCH, 512, 0, stream>>>(inp, hidden, bc, ws_wch, ws_w, ws_u, out);
}
